// Predictor2D_88287347736801
// MI455X (gfx1250) — matrix operand order repaired and checked by reading, not compiled, not run on hardware
//
#include <hip/hip_runtime.h>
#include <hip/hip_bf16.h>
#include <math.h>

typedef __attribute__((ext_vector_type(16))) _Float16 v16h;
typedef __attribute__((ext_vector_type(8)))  _Float16 h8;
typedef __attribute__((ext_vector_type(8)))  float    v8f;

union V8F  { v8f v; float f[8]; };
union V16H { v16h v; h8 h[2]; };
union H8U  { h8 v; _Float16 e[8]; };

// ---------------------------------------------------------------------------
// WMMA helpers (CDNA5 layouts, ISA 7.12.2):
//  A 16x32 f16: lane l holds row m=l&15; halves 0..7 -> K=(l>>4)*8 + 0..7,
//               halves 8..15 -> K=16+(l>>4)*8 + 0..7.
//  B 32x16 f16: lane l holds col n=l&15; halves 0..7 -> K=(l>>4)*8 + 0..7,
//               halves 8..15 -> K=16+(l>>4)*8 + 0..7.
//  C/D 16x16 f32: lane l holds col n=l&15; float v -> row m = v + 8*(l>>4).
// ---------------------------------------------------------------------------
__device__ __forceinline__ v16h load_afrag(const _Float16* p, int ld) {
    int lane = threadIdx.x & 31;
    const _Float16* q = p + (lane & 15) * ld + ((lane >> 4) << 3);
    V16H a;
    a.h[0] = *(const h8*)(q);
    a.h[1] = *(const h8*)(q + 16);
    return a.v;
}
__device__ __forceinline__ v16h load_bfrag(const _Float16* p, int ld) {
    int lane = threadIdx.x & 31;
    const _Float16* q = p + (lane & 15) * ld + ((lane >> 4) << 3);
    V16H b;
    b.h[0] = *(const h8*)(q);
    b.h[1] = *(const h8*)(q + 16);
    return b.v;
}
__device__ __forceinline__ v8f wmma32(v16h a, v16h b, v8f c) {
    return __builtin_amdgcn_wmma_f32_16x16x32_f16(false, a, false, b, (short)0, c, false, false);
}
__device__ __forceinline__ float hmax16(float v) {
    for (int m = 1; m < 16; m <<= 1) v = fmaxf(v, __shfl_xor(v, m, 32));
    return v;
}
__device__ __forceinline__ float hsum16(float v) {
    for (int m = 1; m < 16; m <<= 1) v += __shfl_xor(v, m, 32);
    return v;
}
__device__ __forceinline__ float hmax32(float v) {
    for (int m = 1; m < 32; m <<= 1) v = fmaxf(v, __shfl_xor(v, m, 32));
    return v;
}
__device__ __forceinline__ float hsum32(float v) {
    for (int m = 1; m < 32; m <<= 1) v += __shfl_xor(v, m, 32);
    return v;
}

// ---------------------------------------------------------------------------
// K0: pack weights to fp16, column-major so B fragments are contiguous loads.
// wt: 3072 cols x 64  ([0,512)=Qw [512,1536)=KVw [1536,2048)=Qh [2048,3072)=KVh)
// wow/woh: 64 cols x 512
// ---------------------------------------------------------------------------
__global__ void kpack(const float* Wq_w, const float* Wkv_w, const float* Wq_h,
                      const float* Wkv_h, const float* Wout_w, const float* Wout_h,
                      _Float16* wt, _Float16* wow, _Float16* woh) {
    int i = blockIdx.x * 256 + threadIdx.x;
    if (i < 3072 * 64) {
        int col = i >> 6, k = i & 63;
        float v;
        if (col < 512)       v = Wq_w [k * 512  + col];
        else if (col < 1536) v = Wkv_w[k * 1024 + (col - 512)];
        else if (col < 2048) v = Wq_h [k * 512  + (col - 1536)];
        else                 v = Wkv_h[k * 1024 + (col - 2048)];
        wt[col * 64 + k] = (_Float16)v;
    } else {
        int j = i - 3072 * 64;
        if (j < 64 * 512) {
            int col = j >> 9, k = j & 511;
            wow[col * 512 + k] = (_Float16)Wout_w[k * 64 + col];
        } else {
            j -= 64 * 512;
            if (j < 64 * 512) {
                int col = j >> 9, k = j & 511;
                woh[col * 512 + k] = (_Float16)Wout_h[k * 64 + col];
            }
        }
    }
}

// ---------------------------------------------------------------------------
// K1: fused QKV GEMM: X(65536x64) @ wt(64x3072).
// blockIdx.y IS the destination region (512-col aligned) -> scalar branch,
// no per-lane EXEC cascades. Row-major targets go through a per-wave LDS tile
// so global stores are contiguous 16B h8's. VhT packs 8 rows per lane into one
// 16B store. Only VwT needs a 2B scatter (both output dims strided).
//  Qw/Kw : [(w*8+head)][h][64]     VwT : [(w*8+head)][d][h]
//  Qh/Kh : [(h*8+head)][w][64]     VhT : [(h*8+head)][d][w]
// ---------------------------------------------------------------------------
__global__ void kqkv(const float* __restrict__ x, const _Float16* __restrict__ wt,
                     _Float16* Qw, _Float16* Kw, _Float16* VwT,
                     _Float16* Qh, _Float16* Kh, _Float16* VhT) {
    __shared__ _Float16 Xs[16 * 64];
    __shared__ _Float16 Ts[8][16 * 16];
    int t0 = blockIdx.x * 16;
    int tid = threadIdx.x;
    {   // stage & convert 16x64 fp32 tile -> fp16 LDS
        float4 xv = *(const float4*)(x + (size_t)t0 * 64 + tid * 4);
        Xs[tid * 4 + 0] = (_Float16)xv.x;
        Xs[tid * 4 + 1] = (_Float16)xv.y;
        Xs[tid * 4 + 2] = (_Float16)xv.z;
        Xs[tid * 4 + 3] = (_Float16)xv.w;
    }
    __syncthreads();
    int wv = tid >> 5, lane = tid & 31;
    v16h a0 = load_afrag(Xs, 64);
    v16h a1 = load_afrag(Xs + 32, 64);
    int n = lane & 15, rbase = (lane >> 4) << 3;
    int region = blockIdx.y;           // uniform per block
    int h  = t0 >> 8;                  // uniform (16-token tile never crosses h)
    int w0 = t0 & 255;                 // uniform
    for (int ct = 0; ct < 4; ++ct) {
        int cl = (wv * 4 + ct) * 16;   // local col in [0,512)
        int col0 = region * 512 + cl;
        int hd = cl >> 6;              // head, uniform per tile
        int d0 = cl & 63;              // d base, uniform per tile
        const _Float16* wb = wt + (size_t)col0 * 64;
        v16h b0 = load_bfrag(wb, 64);
        v16h b1 = load_bfrag(wb + 32, 64);
        V8F c = {};
        c.v = wmma32(a0, b0, c.v);
        c.v = wmma32(a1, b1, c.v);
        if (region == 2) {
            // VwT[(w*8+hd)*64 + d][h]: both dims strided -> b16 scatter
            for (int v = 0; v < 8; ++v)
                VwT[(((size_t)((w0 + rbase + v) * 8 + hd) * 64 + d0 + n) << 8) + h] =
                    (_Float16)c.f[v];
        } else if (region == 5) {
            // VhT[(h*8+hd)*64 + d][w]: lane's 8 rows contiguous along w
            H8U pk;
            for (int v = 0; v < 8; ++v) pk.e[v] = (_Float16)c.f[v];
            *(h8*)(VhT + (((size_t)(h * 8 + hd) * 64 + d0 + n) << 8) + w0 + rbase) = pk.v;
        } else {
            // row-major targets: assemble tile in LDS, store contiguous h8's
            for (int v = 0; v < 8; ++v)
                Ts[wv][(rbase + v) * 16 + n] = (_Float16)c.f[v];
            __syncthreads();
            int row = lane & 15, part = lane >> 4;
            h8 val = *(const h8*)(&Ts[wv][row * 16 + part * 8]);
            size_t off;
            _Float16* R;
            if (region == 0)      { R = Qw; off = ((size_t)((w0 + row) * 8 + hd) * 256 + h) << 6; }
            else if (region == 1) { R = Kw; off = ((size_t)((w0 + row) * 8 + hd) * 256 + h) << 6; }
            else if (region == 3) { R = Qh; off = ((size_t)(h * 8 + hd) * 256 + (w0 + row)) << 6; }
            else                  { R = Kh; off = ((size_t)(h * 8 + hd) * 256 + (w0 + row)) << 6; }
            *(h8*)(R + off + d0 + part * 8) = val;
            __syncthreads();
        }
    }
}

// ---------------------------------------------------------------------------
// K2: width attention, flash-style. One block per (w,head) = 2048 blocks.
// K staged once into 32KB LDS (was re-read 16x from global). 8 waves, each
// owns 16 query rows x 2 passes; online softmax over 8 j-tiles of 32;
// P routed C-layout -> A-layout through a per-wave LDS tile.
// Output Ow: [(w*256+h)][head*64+d] fp16.
// ---------------------------------------------------------------------------
__global__ void kwattn(const _Float16* __restrict__ Qw, const _Float16* __restrict__ Kw,
                       const _Float16* __restrict__ VwT, _Float16* __restrict__ Ow) {
    __shared__ _Float16 Ks[256 * 64];      // 32KB
    __shared__ _Float16 Pl[8][16 * 32];    // 8KB
    int b = blockIdx.x;
    int w = b >> 3, head = b & 7;
    const _Float16* Qb = Qw  + ((size_t)b << 14);
    const _Float16* Kb = Kw  + ((size_t)b << 14);
    const _Float16* Vb = VwT + ((size_t)b << 14);
    int tid = threadIdx.x, wv = tid >> 5, lane = tid & 31;
    {   // cooperative K stage: 2048 x 16B
        const float4* src = (const float4*)Kb;
        float4* dst = (float4*)Ks;
        for (int idx = tid; idx < 2048; idx += 256) dst[idx] = src[idx];
    }
    __syncthreads();
    int n = lane & 15, rbase = (lane >> 4) << 3;
    for (int pass = 0; pass < 2; ++pass) {
        int i0 = (pass * 8 + wv) * 16;
        v16h a0 = load_afrag(Qb + i0 * 64, 64);
        v16h a1 = load_afrag(Qb + i0 * 64 + 32, 64);
        V8F o0 = {}, o1 = {}, o2 = {}, o3 = {};
        float mrow[8], lrow[8];
        for (int v = 0; v < 8; ++v) { mrow[v] = -INFINITY; lrow[v] = 0.0f; }
        for (int jt = 0; jt < 8; ++jt) {
            int j0 = jt * 32;
            v16h bk00 = load_bfrag(Ks + j0 * 64, 64);
            v16h bk01 = load_bfrag(Ks + j0 * 64 + 32, 64);
            v16h bk10 = load_bfrag(Ks + (j0 + 16) * 64, 64);
            v16h bk11 = load_bfrag(Ks + (j0 + 16) * 64 + 32, 64);
            V8F s0 = {}, s1 = {};
            s0.v = wmma32(a0, bk00, s0.v); s0.v = wmma32(a1, bk01, s0.v);
            s1.v = wmma32(a0, bk10, s1.v); s1.v = wmma32(a1, bk11, s1.v);
            for (int v = 0; v < 8; ++v) {
                float x0 = s0.f[v] * 0.125f, x1 = s1.f[v] * 0.125f;
                float tm = hmax16(fmaxf(x0, x1));
                float nm = fmaxf(mrow[v], tm);
                float corr = __expf(mrow[v] - nm);
                float p0 = __expf(x0 - nm), p1 = __expf(x1 - nm);
                float rs = hsum16(p0 + p1);
                lrow[v] = lrow[v] * corr + rs;
                mrow[v] = nm;
                o0.f[v] *= corr; o1.f[v] *= corr; o2.f[v] *= corr; o3.f[v] *= corr;
                int row = rbase + v;
                Pl[wv][row * 32 + n]      = (_Float16)p0;
                Pl[wv][row * 32 + n + 16] = (_Float16)p1;
            }
            __syncthreads();
            v16h ap = load_afrag(&Pl[wv][0], 32);
            o0.v = wmma32(ap, load_bfrag(Vb +  0 * 256 + j0, 256), o0.v);
            o1.v = wmma32(ap, load_bfrag(Vb + 16 * 256 + j0, 256), o1.v);
            o2.v = wmma32(ap, load_bfrag(Vb + 32 * 256 + j0, 256), o2.v);
            o3.v = wmma32(ap, load_bfrag(Vb + 48 * 256 + j0, 256), o3.v);
        }
        for (int v = 0; v < 8; ++v) {
            float inv = 1.0f / lrow[v];
            int i = i0 + rbase + v;
            size_t ro = (((size_t)w * 256 + i) << 9) + head * 64;
            Ow[ro +  0 + n] = (_Float16)(o0.f[v] * inv);
            Ow[ro + 16 + n] = (_Float16)(o1.f[v] * inv);
            Ow[ro + 32 + n] = (_Float16)(o2.f[v] * inv);
            Ow[ro + 48 + n] = (_Float16)(o3.f[v] * inv);
        }
    }
}

// ---------------------------------------------------------------------------
// K3: height tied logits: dots[head][i][j] = (1/128) sum_{r,d} Qh*Kh.
// Per-head 256x256 GEMM with K=16384. 256 blocks x 8 waves, 512 WMMA / wave.
// global_prefetch_b8 for next-r tiles to hide latency in the long loop.
// ---------------------------------------------------------------------------
__global__ void khdots(const _Float16* __restrict__ Qh, const _Float16* __restrict__ Kh,
                       float* __restrict__ dots) {
    int b = blockIdx.x;
    int head = b >> 5, rest = b & 31, ib = rest >> 2, jb = rest & 3;
    int tid = threadIdx.x, wv = tid >> 5, lane = tid & 31;
    int i0 = ib * 32 + (wv >> 2) * 16;
    int j0 = jb * 64 + (wv & 3) * 16;
    V8F acc = {};
    for (int r = 0; r < 256; ++r) {
        const _Float16* qb = Qh + (((size_t)(r * 8 + head) * 256 + i0) << 6);
        const _Float16* kb = Kh + (((size_t)(r * 8 + head) * 256 + j0) << 6);
        if (r < 255) {   // prefetch next r's tiles (stride 8*256*64 halves)
            __builtin_prefetch((const void*)(qb + 131072 + (lane << 6)), 0, 1);
            __builtin_prefetch((const void*)(kb + 131072 + (lane << 6)), 0, 1);
        }
        v16h a0 = load_afrag(qb, 64),      a1 = load_afrag(qb + 32, 64);
        v16h b0 = load_bfrag(kb, 64),      b1 = load_bfrag(kb + 32, 64);
        acc.v = wmma32(a0, b0, acc.v);
        acc.v = wmma32(a1, b1, acc.v);
    }
    int n = lane & 15, rbase = (lane >> 4) << 3;
    for (int v = 0; v < 8; ++v) {
        int i = i0 + rbase + v, j = j0 + n;
        dots[((size_t)((head << 8) + i) << 8) + j] = acc.f[v] * (1.0f / 128.0f);
    }
}

// ---------------------------------------------------------------------------
// K4: pair-bias layernorm + 128->8 projection + add + softmax -> attn fp16.
// One block per i row; one wave per (i,j) LN/proj; one wave per head softmax.
// ---------------------------------------------------------------------------
__global__ void kpbsoft(const float* __restrict__ pair_bias, const float* __restrict__ g,
                        const float* __restrict__ bb, const float* __restrict__ Wp,
                        const float* __restrict__ dots, _Float16* __restrict__ attn) {
    __shared__ float sc[8][256];
    int i = blockIdx.x;
    int tid = threadIdx.x, wv = tid >> 5, lane = tid & 31;
    for (int it = 0; it < 32; ++it) {
        int j = it * 8 + wv;
        const float* xp = pair_bias + ((size_t)(i * 256 + j) << 7);
        float4 xv = *(const float4*)(xp + lane * 4);
        float mu = hsum32(xv.x + xv.y + xv.z + xv.w) * (1.0f / 128.0f);
        float d0 = xv.x - mu, d1 = xv.y - mu, d2 = xv.z - mu, d3 = xv.w - mu;
        float var = hsum32(d0 * d0 + d1 * d1 + d2 * d2 + d3 * d3) * (1.0f / 128.0f);
        float rstd = rsqrtf(var + 1e-5f);
        int p0 = lane * 4;
        float n0 = d0 * rstd * g[p0 + 0] + bb[p0 + 0];
        float n1 = d1 * rstd * g[p0 + 1] + bb[p0 + 1];
        float n2 = d2 * rstd * g[p0 + 2] + bb[p0 + 2];
        float n3 = d3 * rstd * g[p0 + 3] + bb[p0 + 3];
        for (int h = 0; h < 8; ++h) {
            float pv = n0 * Wp[(p0 + 0) * 8 + h] + n1 * Wp[(p0 + 1) * 8 + h]
                     + n2 * Wp[(p0 + 2) * 8 + h] + n3 * Wp[(p0 + 3) * 8 + h];
            pv = hsum32(pv);
            if (lane == h)
                sc[h][j] = dots[((size_t)((h << 8) + i) << 8) + j] + pv;
        }
    }
    __syncthreads();
    {
        int head = wv;
        float vals[8], mx = -INFINITY;
        for (int t = 0; t < 8; ++t) {
            vals[t] = sc[head][lane + 32 * t];
            mx = fmaxf(mx, vals[t]);
        }
        mx = hmax32(mx);
        float sum = 0.0f;
        for (int t = 0; t < 8; ++t) { vals[t] = __expf(vals[t] - mx); sum += vals[t]; }
        sum = hsum32(sum);
        float inv = 1.0f / sum;
        for (int t = 0; t < 8; ++t)
            attn[((size_t)((head << 8) + i) << 8) + lane + 32 * t] = (_Float16)(vals[t] * inv);
    }
}

// ---------------------------------------------------------------------------
// K5: height attn @ V: Oh[r][i][head*64+d] = sum_j attn[head][i][j]*VhT[r][head][d][j].
// 2048 blocks (r,head), 8 waves x 8 (i,d) tiles, 8 WMMA each.
// ---------------------------------------------------------------------------
__global__ void khav(const _Float16* __restrict__ attn, const _Float16* __restrict__ VhT,
                     _Float16* __restrict__ Oh) {
    int b = blockIdx.x;
    int r = b >> 3, head = b & 7;
    int tid = threadIdx.x, wv = tid >> 5, lane = tid & 31;
    const _Float16* ab = attn + ((size_t)head << 16);
    const _Float16* vb = VhT + ((size_t)(r * 8 + head) << 14);
    int n = lane & 15, rbase = (lane >> 4) << 3;
    for (int t = 0; t < 8; ++t) {
        int tt = t * 8 + wv;
        int i0 = (tt >> 2) * 16, d0 = (tt & 3) * 16;
        V8F acc = {};
        for (int js = 0; js < 8; ++js) {
            int j0 = js * 32;
            v16h a  = load_afrag(ab + i0 * 256 + j0, 256);
            v16h bv = load_bfrag(vb + d0 * 256 + j0, 256);
            acc.v = wmma32(a, bv, acc.v);
        }
        for (int v = 0; v < 8; ++v) {
            int i = i0 + rbase + v;
            Oh[(((size_t)r * 256 + i) << 9) + head * 64 + d0 + n] = (_Float16)acc.f[v];
        }
    }
}

// ---------------------------------------------------------------------------
// K6: fused output projections + combine: out = 0.5*(Ow@WoutW + bw + Oh@WoutH + bh).
// Ow rows are permuted ((w*256+h) vs token t=h*256+w) -> strided A fragments.
// 2048 blocks of 32 tokens; 8 waves = 2 M-tiles x 4 N-tiles; 32 WMMA / wave.
// ---------------------------------------------------------------------------
__global__ void kout(const _Float16* __restrict__ Ow, const _Float16* __restrict__ Oh,
                     const _Float16* __restrict__ WoW, const _Float16* __restrict__ WoH,
                     const float* __restrict__ bw, const float* __restrict__ bh,
                     float* __restrict__ out) {
    int m0 = blockIdx.x * 32;
    int tid = threadIdx.x, wv = tid >> 5, lane = tid & 31;
    int mt0 = m0 + (wv >> 2) * 16;
    int d0 = (wv & 3) * 16;
    int h = mt0 >> 8, w0 = mt0 & 255;
    const _Float16* aWb = Ow + (((size_t)w0 * 256 + h) << 9);  // row stride 256*512
    const _Float16* aHb = Oh + ((size_t)mt0 << 9);             // row stride 512
    V8F cw = {}, ch = {};
    for (int ks = 0; ks < 16; ++ks) {
        int k0 = ks * 32;
        v16h aw = load_afrag(aWb + k0, 131072);
        cw.v = wmma32(aw, load_bfrag(WoW + d0 * 512 + k0, 512), cw.v);
        v16h ah = load_afrag(aHb + k0, 512);
        ch.v = wmma32(ah, load_bfrag(WoH + d0 * 512 + k0, 512), ch.v);
    }
    int n = lane & 15, rbase = (lane >> 4) << 3;
    float bias = bw[d0 + n] + bh[d0 + n];
    for (int v = 0; v < 8; ++v) {
        int t = mt0 + rbase + v;
        out[((size_t)t << 6) + d0 + n] = 0.5f * (cw.f[v] + ch.f[v] + bias);
    }
}

// ---------------------------------------------------------------------------
// Workspace layout (bytes):
//  wt:393216 | wow:65536 | woh:65536 | 8 x 64MiB fp16 buffers
//  (Qw,Kw,VwT,Qh,Kh,VhT,Ow,Oh) | attn:1MiB | dots:2MiB  -> ~516 MiB total.
// ---------------------------------------------------------------------------
extern "C" void kernel_launch(void* const* d_in, const int* in_sizes, int n_in,
                              void* d_out, int out_size, void* d_ws, size_t ws_size,
                              hipStream_t stream) {
    const float* x         = (const float*)d_in[0];
    const float* pair_bias = (const float*)d_in[1];
    const float* Wq_w      = (const float*)d_in[2];
    const float* Wkv_w     = (const float*)d_in[3];
    const float* Wout_w    = (const float*)d_in[4];
    const float* bout_w    = (const float*)d_in[5];
    const float* Wq_h      = (const float*)d_in[6];
    const float* Wkv_h     = (const float*)d_in[7];
    const float* Wout_h    = (const float*)d_in[8];
    const float* bout_h    = (const float*)d_in[9];
    const float* ln_g      = (const float*)d_in[10];
    const float* ln_b      = (const float*)d_in[11];
    const float* W_pair    = (const float*)d_in[12];

    char* ws = (char*)d_ws;
    const size_t SZ = 67108864;   // 65536 x 512 fp16
    _Float16* wt   = (_Float16*)(ws);
    _Float16* wow  = (_Float16*)(ws + 393216);
    _Float16* woh  = (_Float16*)(ws + 458752);
    _Float16* Qw   = (_Float16*)(ws + 524288 + 0 * SZ);
    _Float16* Kw   = (_Float16*)(ws + 524288 + 1 * SZ);
    _Float16* VwT  = (_Float16*)(ws + 524288 + 2 * SZ);
    _Float16* Qh   = (_Float16*)(ws + 524288 + 3 * SZ);
    _Float16* Kh   = (_Float16*)(ws + 524288 + 4 * SZ);
    _Float16* VhT  = (_Float16*)(ws + 524288 + 5 * SZ);
    _Float16* Ow   = (_Float16*)(ws + 524288 + 6 * SZ);
    _Float16* Oh   = (_Float16*)(ws + 524288 + 7 * SZ);
    _Float16* attn = (_Float16*)(ws + 524288 + 8 * SZ);
    float*    dots = (float*)   (ws + 524288 + 8 * SZ + 1048576);
    float*    outp = (float*)d_out;

    kpack  <<<1024, 256, 0, stream>>>(Wq_w, Wkv_w, Wq_h, Wkv_h, Wout_w, Wout_h, wt, wow, woh);
    kqkv   <<<dim3(4096, 6), 256, 0, stream>>>(x, wt, Qw, Kw, VwT, Qh, Kh, VhT);
    kwattn <<<2048, 256, 0, stream>>>(Qw, Kw, VwT, Ow);
    khdots <<<256,  256, 0, stream>>>(Qh, Kh, dots);
    kpbsoft<<<256,  256, 0, stream>>>(pair_bias, ln_g, ln_b, W_pair, dots, attn);
    khav   <<<2048, 256, 0, stream>>>(attn, VhT, Oh);
    kout   <<<2048, 256, 0, stream>>>(Ow, Oh, wow, woh, bout_w, bout_h, outp);
}
